// ExpHydroM100_80479097192460
// MI455X (gfx1250) — matrix operand order repaired and checked by reading, not compiled, not run on hardware
//
#include <hip/hip_runtime.h>
#include <hip/hip_bf16.h>

// CDNA5 / gfx1250, wave32 WMMA.
typedef __attribute__((ext_vector_type(16))) _Float16 v16h;
typedef __attribute__((ext_vector_type(8)))  _Float16 v8h;
typedef __attribute__((ext_vector_type(8)))  float    v8f;

#define BB 64
#define TT 3653
#define HH 128

// ---- branch-free transcendentals on the hardware TRANS32 units -----------
// v_exp_f32 computes 2^x; v_rcp_f32 is IEEE-ish rcp. All saturate gracefully
// via inf/0 (tanh -> +-1, sigmoid -> 0/1) with no exec-mask branches.
__device__ __forceinline__ float fexp(float x){
  return __builtin_amdgcn_exp2f(x * 1.4426950408889634f);
}
__device__ __forceinline__ float frcp(float x){ return __builtin_amdgcn_rcpf(x); }
__device__ __forceinline__ float ftanh(float x){
  return 1.f - 2.f * frcp(1.f + fexp(2.f * x));
}
__device__ __forceinline__ float fsinh(float x){
  float e = fexp(x);
  return 0.5f * (e - frcp(e));
}
// step(x) = (tanh(5x)+1)*0.5 == sigmoid(10x)
__device__ __forceinline__ float fstep(float x){
  return frcp(1.f + fexp(-10.f * x));
}

union AFrag { v16h v; v8h h[2]; };

// A-matrix 16x32 f16 fragment (ISA 7.12.2): lane l holds row l%16;
// with h1 stored row-major in LDS (stride 128 halves), lane needs halves
// [kb*32 + 8*(l/16) + 0..7] and [+16..23]: two 16B LDS loads.
__device__ __forceinline__ v16h load_a_frag(const _Float16* p){
  AFrag u;
  u.h[0] = *(const v8h*)(p);
  u.h[1] = *(const v8h*)(p + 16);
  return u.v;
}

// Build resident W2 B-fragments for this wave's 16-column N-strip.
// B 32x16 f16 per k-block: lane n -> col = n%16, k = kb*32 + 8*(n/16) + i (i<8), kb*32 + 16 + 8*(n/16) + i-8 (i>=8).
__device__ __forceinline__ void load_b_frags(const float* __restrict__ W2,
                                             int N0, int hi, int lr, v16h* bf){
  #pragma unroll
  for (int kb = 0; kb < 4; ++kb){
    v16h bv;
    #pragma unroll
    for (int i = 0; i < 16; ++i){
      int k = kb*32 + hi*8 + (i < 8 ? i : i + 8);
      bv[i] = (_Float16)W2[k*HH + N0 + lr];
    }
    bf[kb] = bv;
  }
}

// ---------------------------------------------------------------------------
// Kernel 1: batched RK4 scan. 4 workgroups x 16 batches. Per RHS eval the
// hidden GEMM (16x128x128) runs on v_wmma_f32_16x16x32_f16 with W2 resident
// in VGPRs. Writes traj (B,T,2) fp32 to workspace.
// ---------------------------------------------------------------------------
__global__ __launch_bounds__(256) void exphydro_ode(
    const float* __restrict__ in,   // (B,T,5)
    const float* __restrict__ lday, // (B,T)
    const float* __restrict__ W1, const float* __restrict__ b1,
    const float* __restrict__ W2, const float* __restrict__ b2,
    const float* __restrict__ W3, const float* __restrict__ b3,
    float* __restrict__ traj)       // (B,T,2)
{
  __shared__ __align__(16) _Float16 h1s[16*HH];
  __shared__ float h2s[16*HH];
  __shared__ float W1s[4*HH];
  __shared__ float b1s[HH];
  __shared__ float b2s[HH];
  __shared__ float W3s[HH*5];
  __shared__ float b3s[8];
  __shared__ float xbuf[16][4];
  __shared__ float outs[16*5];

  const int tid  = threadIdx.x;
  const int lane = tid & 31, wv = tid >> 5;
  const int hi   = lane >> 4, lr = lane & 15;
  const int N0   = wv * 16;

  for (int i = tid; i < 4*HH; i += 256) W1s[i] = W1[i];
  for (int i = tid; i < HH;   i += 256){ b1s[i] = b1[i]; b2s[i] = b2[i]; }
  for (int i = tid; i < HH*5; i += 256) W3s[i] = W3[i];
  if (tid < 5) b3s[tid] = b3[tid];

  v16h bfrag[4];
  load_b_frags(W2, N0, hi, lr, bfrag);

  const int bg = blockIdx.x * 16 + tid;   // global batch for tid<16
  float y0 = 0.f, y1 = 0.f;
  if (tid < 16){
    const float* inb = in + (size_t)bg * TT * 5;
    y0 = inb[0]; y1 = inb[1];
    traj[((size_t)bg*TT + 0)*2 + 0] = y0;
    traj[((size_t)bg*TT + 0)*2 + 1] = y1;
  }
  __syncthreads();

  for (int t = 0; t < TT-1; ++t){
    float te0=0.f, te1=0.f, p0=0.f, p1=0.f, ld0=0.f, ld1=0.f, hstep=1.f;
    float ks[4][2];
    if (tid < 16){
      const float* inb = in + (size_t)bg * TT * 5;
      te0 = inb[(size_t)t*5 + 2];      p0 = inb[(size_t)t*5 + 3];
      te1 = inb[(size_t)(t+1)*5 + 2];  p1 = inb[(size_t)(t+1)*5 + 3];
      ld0 = lday[(size_t)bg*TT + t];   ld1 = lday[(size_t)bg*TT + t + 1];
      hstep = in[(size_t)(t+1)*5 + 4] - in[(size_t)t*5 + 4]; // t_grid diff
    }
    const float tem = 0.5f*(te0+te1), pm = 0.5f*(p0+p1), ldm = 0.5f*(ld0+ld1);

    #pragma unroll
    for (int st = 0; st < 4; ++st){
      float s0s=0.f, s1s=0.f, tes=0.f, lds_=0.f;
      if (tid < 16){
        const float cy  = (st==0) ? 0.f : (st==3) ? hstep : 0.5f*hstep;
        const float k0p = (st==0) ? 0.f : ks[st-1][0];
        const float k1p = (st==0) ? 0.f : ks[st-1][1];
        s0s = y0 + cy*k0p; s1s = y1 + cy*k1p;
        float ps;
        if (st==0){ tes=te0; ps=p0; lds_=ld0; }
        else if (st==3){ tes=te1; ps=p1; lds_=ld1; }
        else { tes=tem; ps=pm; lds_=ldm; }
        xbuf[tid][0]=s0s; xbuf[tid][1]=s1s; xbuf[tid][2]=tes; xbuf[tid][3]=ps;
      }
      __syncthreads();

      { // layer1: 16x4 @ 4x128, tanh -> h1s (f16, row-major)
        const int b = tid >> 4, j0 = (tid & 15) * 8;
        const float x0=xbuf[b][0], x1=xbuf[b][1], x2=xbuf[b][2], x3=xbuf[b][3];
        #pragma unroll
        for (int j = j0; j < j0+8; ++j){
          float a = b1s[j] + x0*W1s[j] + x1*W1s[HH+j] + x2*W1s[2*HH+j] + x3*W1s[3*HH+j];
          h1s[b*HH + j] = (_Float16)ftanh(a);
        }
      }
      __syncthreads();

      { // layer2: WMMA 16x128x128, tanh -> h2s (f32)
        v8f acc;
        #pragma unroll
        for (int r = 0; r < 8; ++r) acc[r] = 0.f;
        #pragma unroll
        for (int kb = 0; kb < 4; ++kb){
          v16h a = load_a_frag(&h1s[lr*HH + kb*32 + hi*8]);
          acc = __builtin_amdgcn_wmma_f32_16x16x32_f16(
                    false, a, false, bfrag[kb], (short)0, acc, false, false);
        }
        #pragma unroll
        for (int r = 0; r < 8; ++r){
          const int m = r + 8*hi;
          const int n = N0 + lr;
          h2s[m*HH + n] = ftanh(acc[r] + b2s[n]);
        }
      }
      __syncthreads();

      if (tid < 80){ // layer3: 16x128 @ 128x5
        const int b = tid/5, o = tid%5;
        float s = b3s[o];
        #pragma unroll 8
        for (int j = 0; j < HH; ++j) s += h2s[b*HH + j] * W3s[j*5 + o];
        outs[b*5 + o] = s;
      }
      __syncthreads();

      if (tid < 16){ // RHS nonlinearities (branch-free trans ops)
        const float o0=outs[tid*5+0], o1=outs[tid*5+1], o2=outs[tid*5+2];
        const float o3=outs[tid*5+3], o4=outs[tid*5+4];
        const float psnow = fmaxf(0.f, fsinh(o0)*fstep(-tes));
        const float prain = fmaxf(0.f, fsinh(o1));
        const float melt  = fmaxf(0.f, fstep(s0s)*fsinh(o2));
        const float et    = fstep(s1s)*fexp(o3)*lds_;
        const float q     = fstep(s1s)*fexp(o4);
        ks[st][0] = psnow - melt;
        ks[st][1] = prain + melt - et - q;
      }
      __syncthreads();
    }

    if (tid < 16){
      y0 += hstep*(1.f/6.f)*(ks[0][0] + 2.f*ks[1][0] + 2.f*ks[2][0] + ks[3][0]);
      y1 += hstep*(1.f/6.f)*(ks[0][1] + 2.f*ks[1][1] + 2.f*ks[2][1] + ks[3][1]);
      traj[((size_t)bg*TT + t+1)*2 + 0] = y0;
      traj[((size_t)bg*TT + t+1)*2 + 1] = y1;
    }
  }
}

// ---------------------------------------------------------------------------
// Kernel 2: MLP head over all B*T rows (16-row tiles), output column 4 only.
// Same WMMA machinery; fully parallel, grid-stride over 14612 tiles.
// ---------------------------------------------------------------------------
__global__ __launch_bounds__(256) void exphydro_head(
    const float* __restrict__ in,   // (B,T,5)
    const float* __restrict__ traj, // (B,T,2)
    const float* __restrict__ W1, const float* __restrict__ b1,
    const float* __restrict__ W2, const float* __restrict__ b2,
    const float* __restrict__ W3, const float* __restrict__ b3,
    float* __restrict__ out)        // (B,T)
{
  __shared__ __align__(16) _Float16 h1s[16*HH];
  __shared__ float h2s[16*HH];
  __shared__ float W1s[4*HH];
  __shared__ float b1s[HH];
  __shared__ float b2s[HH];
  __shared__ float w3c[HH];
  __shared__ float xbuf[16][4];

  const int tid  = threadIdx.x;
  const int lane = tid & 31, wv = tid >> 5;
  const int hi   = lane >> 4, lr = lane & 15;
  const int N0   = wv * 16;

  for (int i = tid; i < 4*HH; i += 256) W1s[i] = W1[i];
  for (int i = tid; i < HH;   i += 256){ b1s[i]=b1[i]; b2s[i]=b2[i]; w3c[i]=W3[i*5+4]; }
  const float b3c = b3[4];

  v16h bfrag[4];
  load_b_frags(W2, N0, hi, lr, bfrag);
  __syncthreads();

  const int NTILES = (BB*TT)/16;   // 233792/16 = 14612 exactly
  for (int tile = blockIdx.x; tile < NTILES; tile += gridDim.x){
    if (tid < 16){
      const size_t rr = (size_t)tile*16 + tid;   // rr = b*T + t
      xbuf[tid][0] = fmaxf(0.f, traj[rr*2 + 0]);
      xbuf[tid][1] = fmaxf(0.f, traj[rr*2 + 1]);
      xbuf[tid][2] = in[rr*5 + 2];
      xbuf[tid][3] = in[rr*5 + 3];
    }
    __syncthreads();

    { // layer1
      const int b = tid >> 4, j0 = (tid & 15) * 8;
      const float x0=xbuf[b][0], x1=xbuf[b][1], x2=xbuf[b][2], x3=xbuf[b][3];
      #pragma unroll
      for (int j = j0; j < j0+8; ++j){
        float a = b1s[j] + x0*W1s[j] + x1*W1s[HH+j] + x2*W1s[2*HH+j] + x3*W1s[3*HH+j];
        h1s[b*HH + j] = (_Float16)ftanh(a);
      }
    }
    __syncthreads();

    { // layer2 WMMA
      v8f acc;
      #pragma unroll
      for (int r = 0; r < 8; ++r) acc[r] = 0.f;
      #pragma unroll
      for (int kb = 0; kb < 4; ++kb){
        v16h a = load_a_frag(&h1s[lr*HH + kb*32 + hi*8]);
        acc = __builtin_amdgcn_wmma_f32_16x16x32_f16(
                  false, a, false, bfrag[kb], (short)0, acc, false, false);
      }
      #pragma unroll
      for (int r = 0; r < 8; ++r){
        const int m = r + 8*hi;
        const int n = N0 + lr;
        h2s[m*HH + n] = ftanh(acc[r] + b2s[n]);
      }
    }
    __syncthreads();

    if (tid < 16){ // layer3: only output column 4
      float s = b3c;
      #pragma unroll 8
      for (int j = 0; j < HH; ++j) s += h2s[tid*HH + j] * w3c[j];
      out[(size_t)tile*16 + tid] = s;
    }
    __syncthreads();
  }
}

extern "C" void kernel_launch(void* const* d_in, const int* in_sizes, int n_in,
                              void* d_out, int out_size, void* d_ws, size_t ws_size,
                              hipStream_t stream) {
  (void)in_sizes; (void)n_in; (void)out_size; (void)ws_size;
  const float* in  = (const float*)d_in[0];   // (B,T,5)
  const float* ld  = (const float*)d_in[1];   // (B,T)
  const float* W1  = (const float*)d_in[2];
  const float* b1  = (const float*)d_in[3];
  const float* W2  = (const float*)d_in[4];
  const float* b2  = (const float*)d_in[5];
  const float* W3  = (const float*)d_in[6];
  const float* b3  = (const float*)d_in[7];
  float* traj = (float*)d_ws;                 // B*T*2 fp32 = 1.87 MB
  float* out  = (float*)d_out;                // (B,T) fp32

  hipLaunchKernelGGL(exphydro_ode, dim3(BB/16), dim3(256), 0, stream,
                     in, ld, W1, b1, W2, b2, W3, b3, traj);
  hipLaunchKernelGGL(exphydro_head, dim3(1024), dim3(256), 0, stream,
                     in, traj, W1, b1, W2, b2, W3, b3, out);
}
